// PhotonicAttentionMechanism_77068893160347
// MI455X (gfx1250) — matrix operand order repaired and checked by reading, not compiled, not run on hardware
//
#include <hip/hip_runtime.h>
#include <hip/hip_bf16.h>

// ---------------- WMMA fragment types (CDNA5 wave32) ----------------
typedef __attribute__((ext_vector_type(16))) __bf16 v16bf;
typedef __attribute__((ext_vector_type(8)))  __bf16 v8bf;
typedef __attribute__((ext_vector_type(8)))  float  v8f;
typedef __attribute__((ext_vector_type(4)))  int    v4i;

#define EMBED 1024
#define SEQ   1024
#define NH    16
#define HD    64
#define WL    16
#define KERRC 2.7e-12f     // 2.7e-18 * 1e6
#define SCALE 0.125f       // 1/sqrt(64)

// workspace layout (byte offsets; weights deliberately at constant spacing so
// device code can use one base pointer + immediate offsets -> no pointer spills)
#define MB        ((size_t)1 << 20)
#define OFF_XB    (0 * MB)
#define OFF_WQ    (2 * MB)
#define OFF_WK    (4 * MB)
#define OFF_WV    (6 * MB)
#define OFF_WO    (8 * MB)
#define OFF_Q     (10 * MB)
#define OFF_KT    (12 * MB)
#define OFF_V     (14 * MB)
#define OFF_COMB  (16 * MB)
#define WSTRIDE_E ((size_t)1 << 20)   // element stride between adjacent weight buffers

#if __has_builtin(__builtin_amdgcn_global_load_async_to_lds_b128)
#define HAVE_ASYNC_LDS 1
#else
#define HAVE_ASYNC_LDS 0
#endif

typedef __attribute__((address_space(1))) v4i* gv4i_p;
typedef __attribute__((address_space(3))) v4i* lv4i_p;

static __device__ inline v8f vzero() {
  v8f z;
#pragma unroll
  for (int i = 0; i < 8; ++i) z[i] = 0.0f;
  return z;
}
static __device__ inline __bf16 f2bf(float f) { return (__bf16)f; }

static __device__ inline v16bf mk16(v8bf lo, v8bf hi) {
  v16bf r;
#pragma unroll
  for (int i = 0; i < 8; ++i) { r[i] = lo[i]; r[i + 8] = hi[i]; }
  return r;
}

// A fragment (16x32 bf16, M x K): lane = M row; lanes<16 hold K {0..7,16..23},
// lanes>=16 hold K {8..15,24..31} of the 32-wide k-step.
static __device__ inline v16bf loadA(const __bf16* src, int ld, int row0, int k0, int lane) {
  const int m  = row0 + (lane & 15);
  const int ko = k0 + ((lane >> 4) << 3);
  v8bf lo = *(const v8bf*)(src + (size_t)m * ld + ko);
  v8bf hi = *(const v8bf*)(src + (size_t)m * ld + ko + 16);
  return mk16(lo, hi);
}

// B fragment (32x16 bf16, K x N): lane = N col; lanes<16 hold K {0..7,16..23},
// lanes>=16 hold K {8..15,24..31} of the 32-wide k-step.
static __device__ inline v16bf loadB(const __bf16* src, int ld, int k0, int n0, int lane) {
  const __bf16* p = src + (size_t)(k0 + ((lane >> 4) << 3)) * ld + n0 + (lane & 15);
  v16bf r;
#pragma unroll
  for (int i = 0; i < 8; ++i) { r[i] = p[(size_t)i * ld]; r[i + 8] = p[(size_t)(i + 16) * ld]; }
  return r;
}

static __device__ inline v8f wmma_bf16(v16bf a, v16bf b, v8f c) {
  return __builtin_amdgcn_wmma_f32_16x16x32_bf16(false, a, false, b, (short)0, c, false, false);
}

static __device__ inline void wait_asynccnt0() {
#if __has_builtin(__builtin_amdgcn_s_wait_asynccnt)
  __builtin_amdgcn_s_wait_asynccnt(0);
#else
  asm volatile("s_wait_asynccnt 0" ::: "memory");
#endif
}

static __device__ inline void cp_lds16(const __bf16* src, __bf16* dst) {
#if HAVE_ASYNC_LDS
  __builtin_amdgcn_global_load_async_to_lds_b128((gv4i_p)src, (lv4i_p)dst, 0, 0);
#else
  *(v8bf*)dst = *(const v8bf*)src;
#endif
}

// ---------------- fp32 -> bf16 convert ----------------
__global__ void cvt_kernel(const float* __restrict__ src, __bf16* __restrict__ dst, int n) {
  for (int i = blockIdx.x * blockDim.x + threadIdx.x; i < n; i += gridDim.x * blockDim.x)
    dst[i] = f2bf(src[i]);
}

// ---------------- fused QKV projection GEMM ----------------
// Block = 128 rows x 64 cols; all 8 waves share the per-k-step B tiles, which
// are double-buffer staged into LDS with async-to-LDS DMA.
__global__ void qkv_gemm_kernel(const char* __restrict__ wsb,
                                const float* __restrict__ bq,
                                const float* __restrict__ bk,
                                const float* __restrict__ bv) {
  const __bf16* xb = (const __bf16*)(wsb + OFF_XB);
  const __bf16* wq = (const __bf16*)(wsb + OFF_WQ);   // wk = wq+1M elems, wv = wq+2M
  __bf16* Qb = (__bf16*)(wsb + OFF_Q);
  __bf16* Kt = (__bf16*)(wsb + OFF_KT);
  __bf16* Vb = (__bf16*)(wsb + OFF_V);

  __shared__ __bf16 Blds[2 * 3 * 32 * 64];            // 24KB double-buffered B stage

  const int lane = threadIdx.x & 31;
  const int wave = threadIdx.x >> 5;
  const int row0 = (blockIdx.x >> 4) * 128 + wave * 16;
  const int col0 = (blockIdx.x & 15) * 64;

  // stage B tiles (3 weights x 32x64) for k-step ks into buffer buf
  auto stage = [&](int ks, int buf) {
#pragma unroll
    for (int i = 0; i < 3; ++i) {
      const int idx = threadIdx.x + i * 256;          // 768 x 16B chunks
      const int w   = idx >> 8;
      const int kr  = (idx >> 3) & 31;
      const int c8  = idx & 7;
      cp_lds16(wq + (size_t)w * WSTRIDE_E + (size_t)(ks * 32 + kr) * EMBED + col0 + c8 * 8,
               Blds + buf * 6144 + w * 2048 + kr * 64 + c8 * 8);
    }
  };

  v8f cq[4], ck[4], cv[4];
#pragma unroll
  for (int nt = 0; nt < 4; ++nt) { cq[nt] = vzero(); ck[nt] = vzero(); cv[nt] = vzero(); }

  stage(0, 0);
#if HAVE_ASYNC_LDS
  wait_asynccnt0();
#endif
  __syncthreads();

  for (int ks = 0; ks < 32; ++ks) {
    const int buf = ks & 1;
    if (ks < 31) stage(ks + 1, buf ^ 1);              // DMA next tile during compute
    const __bf16* cur = Blds + buf * 6144;
    v16bf a = loadA(xb, EMBED, row0, ks * 32, lane);
#pragma unroll
    for (int nt = 0; nt < 4; ++nt) {
      v16bf b0 = loadB(cur + 0 * 2048, 64, 0, nt * 16, lane);
      v16bf b1 = loadB(cur + 1 * 2048, 64, 0, nt * 16, lane);
      v16bf b2 = loadB(cur + 2 * 2048, 64, 0, nt * 16, lane);
      cq[nt] = wmma_bf16(a, b0, cq[nt]);
      ck[nt] = wmma_bf16(a, b1, ck[nt]);
      cv[nt] = wmma_bf16(a, b2, cv[nt]);
    }
#if HAVE_ASYNC_LDS
    wait_asynccnt0();
#endif
    __syncthreads();
  }

#pragma unroll
  for (int nt = 0; nt < 4; ++nt) {
    const int col = col0 + nt * 16 + (lane & 15);
    const float biq = bq[col], bik = bk[col], biv = bv[col];
#pragma unroll
    for (int r = 0; r < 8; ++r) {
      const int row = row0 + r + ((lane >> 4) << 3);
      Qb[(size_t)row * EMBED + col] = f2bf(cq[nt][r] + biq);
      Vb[(size_t)row * EMBED + col] = f2bf(cv[nt][r] + biv);
      Kt[(size_t)col * SEQ  + row] = f2bf(ck[nt][r] + bik);
    }
  }
}

// ---------------- multi-channel flash attention ----------------
// Block = (head h, 128-query group). LDS: Kt_h 64x1024 (128KB) + V_h 1024x64
// (128KB) + 8 per-wave P staging buffers 16x128 (32KB) = 288KB of the 320KB WGP LDS.
__global__ void attn_kernel(const char* __restrict__ wsb,
                            const float* __restrict__ pm,   // [WL][NH]
                            const float* __restrict__ iwv) {// [WL]
  const __bf16* Qb = (const __bf16*)(wsb + OFF_Q);
  const __bf16* Kt = (const __bf16*)(wsb + OFF_KT);
  const __bf16* Vb = (const __bf16*)(wsb + OFF_V);
  __bf16* comb = (__bf16*)(wsb + OFF_COMB);

  extern __shared__ __align__(16) char smem[];
  __bf16* Klds = (__bf16*)smem;                       // [64][1024]
  __bf16* Vlds = (__bf16*)(smem + 131072);            // [1024][64]
  const int lane = threadIdx.x & 31;
  const int wave = threadIdx.x >> 5;
  __bf16* Pst = (__bf16*)(smem + 262144) + wave * 2048;  // [16][128] per wave

  const int h    = blockIdx.x >> 3;
  const int row0 = (blockIdx.x & 7) * 128 + wave * 16;

  for (int it = 0; it < 32; ++it) {
    const int idx = it * 256 + threadIdx.x;
    { const int r = idx >> 7, c = idx & 127;   // Klds: 64 rows x 128 chunks
      cp_lds16(Kt + (size_t)(h * HD + r) * SEQ + c * 8, Klds + r * 1024 + c * 8); }
    { const int r = idx >> 3, c = idx & 7;     // Vlds: 1024 rows x 8 chunks
      cp_lds16(Vb + (size_t)r * EMBED + h * HD + c * 8, Vlds + r * 64 + c * 8); }
  }
#if HAVE_ASYNC_LDS
  wait_asynccnt0();
#endif
  __syncthreads();

  // Q A-fragments for this wave's 16 rows (reused across all channels)
  v16bf aQ0 = loadA(Qb, EMBED, row0, h * HD + 0, lane);
  v16bf aQ1 = loadA(Qb, EMBED, row0, h * HD + 32, lane);

  v8f combacc[4];
#pragma unroll
  for (int nt = 0; nt < 4; ++nt) combacc[nt] = vzero();

  for (int wl = 0; wl < WL; ++wl) {
    const float m    = __cosf(pm[wl * NH + h]);
    const float sfac = SCALE * m * m;
    const float iw   = iwv[wl];

    float rmax[8], rsum[8];
    v8f oacc[4];
#pragma unroll
    for (int r = 0; r < 8; ++r) { rmax[r] = -3.0e38f; rsum[r] = 0.0f; }
#pragma unroll
    for (int nt = 0; nt < 4; ++nt) oacc[nt] = vzero();

    for (int kb = 0; kb < 8; ++kb) {      // 128 keys per block
      const int n0 = kb * 128;
      // scores: 8 tiles of Q(16x64) x K^T(64x16), 2-stage pipelined B loads
      v8f sf[8];
      v16bf kb0 = loadB(Klds, 1024, 0,  n0, lane);
      v16bf kb1 = loadB(Klds, 1024, 32, n0, lane);
#pragma unroll
      for (int nt = 0; nt < 8; ++nt) {
        v16bf nb0 = kb0, nb1 = kb1;
        if (nt < 7) {
          nb0 = loadB(Klds, 1024, 0,  n0 + (nt + 1) * 16, lane);
          nb1 = loadB(Klds, 1024, 32, n0 + (nt + 1) * 16, lane);
        }
        v8f c = vzero();
        c = wmma_bf16(aQ0, kb0, c);
        c = wmma_bf16(aQ1, kb1, c);
        sf[nt] = c;
        kb0 = nb0; kb1 = nb1;
      }
      // scale + Kerr nonlinearity + row max
      float nmax[8];
#pragma unroll
      for (int r = 0; r < 8; ++r) nmax[r] = rmax[r];
#pragma unroll
      for (int nt = 0; nt < 8; ++nt)
#pragma unroll
        for (int r = 0; r < 8; ++r) {
          float s0 = sf[nt][r] * sfac;
          float s  = s0 * __cosf(KERRC * s0 * s0);
          sf[nt][r] = s;
          nmax[r] = fmaxf(nmax[r], s);
        }
#pragma unroll
      for (int r = 0; r < 8; ++r) {        // reduce across 16-lane halves
        nmax[r] = fmaxf(nmax[r], __shfl_xor(nmax[r], 1));
        nmax[r] = fmaxf(nmax[r], __shfl_xor(nmax[r], 2));
        nmax[r] = fmaxf(nmax[r], __shfl_xor(nmax[r], 4));
        nmax[r] = fmaxf(nmax[r], __shfl_xor(nmax[r], 8));
      }
#pragma unroll
      for (int r = 0; r < 8; ++r) {
        const float corr = __expf(rmax[r] - nmax[r]);
        rmax[r] = nmax[r];
        rsum[r] *= corr;
#pragma unroll
        for (int nt = 0; nt < 4; ++nt) oacc[nt][r] *= corr;
      }
      // exp, accumulate row sums, stage P (bf16) in LDS in row-major A order
#pragma unroll
      for (int nt = 0; nt < 8; ++nt)
#pragma unroll
        for (int r = 0; r < 8; ++r) {
          const float p = __expf(sf[nt][r] - rmax[r]);
          rsum[r] += p;
          Pst[(r + ((lane >> 4) << 3)) * 128 + nt * 16 + (lane & 15)] = f2bf(p);
        }
      asm volatile("s_wait_dscnt 0" ::: "memory");
      // P(16x128) @ V(128x64), 2-stage pipelined A/B loads
      v16bf pa  = loadA(Pst, 128, 0, 0, lane);
      v16bf vb0 = loadB(Vlds, 64, n0, 0, lane);
#pragma unroll
      for (int k2 = 0; k2 < 4; ++k2) {
        v16bf npa = pa;
        if (k2 < 3) npa = loadA(Pst, 128, 0, (k2 + 1) * 32, lane);
#pragma unroll
        for (int nt = 0; nt < 4; ++nt) {
          v16bf nvb = vb0;
          if (nt < 3)      nvb = loadB(Vlds, 64, n0 + k2 * 32, (nt + 1) * 16, lane);
          else if (k2 < 3) nvb = loadB(Vlds, 64, n0 + (k2 + 1) * 32, 0, lane);
          oacc[nt] = wmma_bf16(pa, vb0, oacc[nt]);
          vb0 = nvb;
        }
        pa = npa;
      }
    }
    // finalize this channel
#pragma unroll
    for (int r = 0; r < 8; ++r) {
      rsum[r] += __shfl_xor(rsum[r], 1);
      rsum[r] += __shfl_xor(rsum[r], 2);
      rsum[r] += __shfl_xor(rsum[r], 4);
      rsum[r] += __shfl_xor(rsum[r], 8);
      const float inv = iw / rsum[r];
#pragma unroll
      for (int nt = 0; nt < 4; ++nt) combacc[nt][r] += oacc[nt][r] * inv;
    }
  }
#pragma unroll
  for (int nt = 0; nt < 4; ++nt)
#pragma unroll
    for (int r = 0; r < 8; ++r) {
      const int row = row0 + r + ((lane >> 4) << 3);
      const int col = h * HD + nt * 16 + (lane & 15);
      comb[(size_t)row * EMBED + col] = f2bf(combacc[nt][r] * (1.0f / WL));
    }
}

// ---------------- output projection GEMM (f32 out) ----------------
// Block = 128 rows x 64 cols, double-buffered async B staging like QKV.
__global__ void out_gemm_kernel(const char* __restrict__ wsb,
                                const float* __restrict__ bo,
                                float* __restrict__ out) {
  const __bf16* xb = (const __bf16*)(wsb + OFF_COMB);
  const __bf16* wo = (const __bf16*)(wsb + OFF_WO);

  __shared__ __bf16 Blds[2 * 32 * 64];                // 8KB double buffer

  const int lane = threadIdx.x & 31;
  const int wave = threadIdx.x >> 5;
  const int row0 = (blockIdx.x >> 4) * 128 + wave * 16;
  const int col0 = (blockIdx.x & 15) * 64;

  auto stage = [&](int ks, int buf) {
    const int idx = threadIdx.x;                      // 256 x 16B chunks
    const int kr  = idx >> 3;
    const int c8  = idx & 7;
    cp_lds16(wo + (size_t)(ks * 32 + kr) * EMBED + col0 + c8 * 8,
             Blds + buf * 2048 + kr * 64 + c8 * 8);
  };

  v8f c[4];
#pragma unroll
  for (int nt = 0; nt < 4; ++nt) c[nt] = vzero();

  stage(0, 0);
#if HAVE_ASYNC_LDS
  wait_asynccnt0();
#endif
  __syncthreads();

  for (int ks = 0; ks < 32; ++ks) {
    const int buf = ks & 1;
    if (ks < 31) stage(ks + 1, buf ^ 1);
    const __bf16* cur = Blds + buf * 2048;
    v16bf a = loadA(xb, EMBED, row0, ks * 32, lane);
#pragma unroll
    for (int nt = 0; nt < 4; ++nt)
      c[nt] = wmma_bf16(a, loadB(cur, 64, 0, nt * 16, lane), c[nt]);
#if HAVE_ASYNC_LDS
    wait_asynccnt0();
#endif
    __syncthreads();
  }

#pragma unroll
  for (int nt = 0; nt < 4; ++nt) {
    const int col = col0 + nt * 16 + (lane & 15);
    const float bias = bo[col];
#pragma unroll
    for (int r = 0; r < 8; ++r) {
      const int row = row0 + r + ((lane >> 4) << 3);
      out[(size_t)row * EMBED + col] = c[nt][r] + bias;
    }
  }
}

// ---------------- host ----------------
extern "C" void kernel_launch(void* const* d_in, const int* in_sizes, int n_in,
                              void* d_out, int out_size, void* d_ws, size_t ws_size,
                              hipStream_t stream) {
  (void)in_sizes; (void)n_in; (void)out_size; (void)ws_size;
  const float* x   = (const float*)d_in[0];
  const float* Wq  = (const float*)d_in[1];
  const float* bq  = (const float*)d_in[2];
  const float* Wk  = (const float*)d_in[3];
  const float* bk  = (const float*)d_in[4];
  const float* Wv  = (const float*)d_in[5];
  const float* bv  = (const float*)d_in[6];
  const float* Wo  = (const float*)d_in[7];
  const float* bo  = (const float*)d_in[8];
  const float* pm  = (const float*)d_in[9];
  const float* iwv = (const float*)d_in[10];

  char* ws = (char*)d_ws;
  const int N = EMBED * SEQ;
  cvt_kernel<<<512, 256, 0, stream>>>(x,  (__bf16*)(ws + OFF_XB), N);
  cvt_kernel<<<512, 256, 0, stream>>>(Wq, (__bf16*)(ws + OFF_WQ), N);
  cvt_kernel<<<512, 256, 0, stream>>>(Wk, (__bf16*)(ws + OFF_WK), N);
  cvt_kernel<<<512, 256, 0, stream>>>(Wv, (__bf16*)(ws + OFF_WV), N);
  cvt_kernel<<<512, 256, 0, stream>>>(Wo, (__bf16*)(ws + OFF_WO), N);

  qkv_gemm_kernel<<<128, 256, 0, stream>>>(ws, bq, bk, bv);

  const int attn_lds = 294912;  // 288KB of the 320KB WGP LDS
  (void)hipFuncSetAttribute((const void*)attn_kernel,
                            hipFuncAttributeMaxDynamicSharedMemorySize, attn_lds);
  attn_kernel<<<NH * 8, 256, attn_lds, stream>>>(ws, pm, iwv);

  out_gemm_kernel<<<128, 256, 0, stream>>>(ws, bo, (float*)d_out);
}
